// RNN_Decoder_17858474017320
// MI455X (gfx1250) — matrix operand order repaired and checked by reading, not compiled, not run on hardware
//
#include <hip/hip_runtime.h>
#include <math.h>

// Sizes from the reference
#define BB 64
#define TT 50
#define EE 512
#define HH 512
#define VV 30000

typedef __bf16 bf16_t;
typedef bf16_t v16bf __attribute__((ext_vector_type(16)));
typedef float  v8f   __attribute__((ext_vector_type(8)));

union FragU { v16bf v; uint4 q[2]; };

__device__ inline unsigned short f2bf(float f) {
    unsigned u = __builtin_bit_cast(unsigned, f);
    unsigned r = u + 0x7FFFu + ((u >> 16) & 1u);   // round-to-nearest-even
    return (unsigned short)(r >> 16);
}

// A fragment: 16x32 bf16 tile from row-major [rows, lda] at (mbase, k0).
// Lane L: row = L%16; K-octets at k0+(L/16)*8 and k0+(L/16)*8+16.
__device__ inline v16bf loadA(const unsigned short* A, int lda, int mbase, int k0, int lane) {
    int r = lane & 15, half = lane >> 4;
    const unsigned short* p = A + (size_t)(mbase + r) * lda + k0 + half * 8;
    FragU f;
    f.q[0] = *(const uint4*)(p);
    f.q[1] = *(const uint4*)(p + 16);
    return f.v;
}

// B fragment: 32x16 bf16 tile B[k,n] = W[n,k] with W row-major [nmax, ldw].
// Lane L: col = L%16; K-octets at k0+(L/16)*8 and k0+(L/16)*8+16 (contiguous in W's row).
__device__ inline v16bf loadB(const unsigned short* W, int ldw, int nbase, int k0, int lane, int nmax) {
    int r = lane & 15, half = lane >> 4;
    int n = nbase + r; if (n >= nmax) n = nmax - 1;   // clamp OOB columns
    const unsigned short* p = W + (size_t)n * ldw + k0 + half * 8;
    FragU f;
    f.q[0] = *(const uint4*)(p);
    f.q[1] = *(const uint4*)(p + 16);
    return f.v;
}

__device__ inline v8f wmma_bf16(v16bf a, v16bf b, v8f c) {
    return __builtin_amdgcn_wmma_f32_16x16x32_bf16(false, a, false, b, (short)0, c, false, false);
}

__device__ inline v8f zero8() {
    v8f z;
    #pragma unroll
    for (int i = 0; i < 8; ++i) z[i] = 0.f;
    return z;
}

__device__ inline float sigm(float x) { return 1.f / (1.f + __expf(-x)); }

// ---------------- f32 -> bf16 convert ----------------
__global__ void cvt_k(const float* __restrict__ src, unsigned short* __restrict__ dst, int n) {
    int i = blockIdx.x * 256 + threadIdx.x;
    if (i < n) dst[i] = f2bf(src[i]);
}

// ---------------- embedding gather -> xs_bf16 [T*B, E] (row = t*64+b) ----------------
__global__ void embed_k(const float* __restrict__ x, const int* __restrict__ caption,
                        const float* __restrict__ emb, unsigned short* __restrict__ xs) {
    int idx = blockIdx.x * 256 + threadIdx.x;
    if (idx >= TT * BB * EE) return;
    int e = idx & (EE - 1);
    int r = idx >> 9;            // t*64 + b
    int t = r >> 6, b = r & 63;
    float v = (t == 0) ? x[b * EE + e]
                       : emb[(size_t)caption[b * TT + t] * EE + e];
    xs[idx] = f2bf(v);
}

// ---------------- one LSTM time step (fused gates GEMM + pointwise) ----------------
// Block nb (0..31) owns H-columns [nb*16, nb*16+16). 8 waves; each wave computes 2 of
// the 16 (gate g, m-tile mm) 16x16 gate tiles, staged in LDS, then pointwise update.
__global__ void lstm_step(const unsigned short* __restrict__ xs, unsigned short* __restrict__ hs,
                          float* __restrict__ cbuf,
                          const unsigned short* __restrict__ Wih, const unsigned short* __restrict__ Whh,
                          const float* __restrict__ bih, const float* __restrict__ bhh, int t) {
    __shared__ float gl[4][BB][16];
    int w = threadIdx.x >> 5, lane = threadIdx.x & 31;
    int n0 = blockIdx.x * 16;

    #pragma unroll
    for (int ti = 0; ti < 2; ++ti) {
        int tileIdx = w * 2 + ti;
        int g = tileIdx >> 2, mm = tileIdx & 3;
        int nbase = g * HH + n0;                 // row in W_ih / W_hh (gate column)
        v8f acc = zero8();
        #pragma unroll 4
        for (int k0 = 0; k0 < EE; k0 += 32)      // gates += x_t @ W_ih^T
            acc = wmma_bf16(loadA(xs, EE, t * BB + mm * 16, k0, lane),
                            loadB(Wih, EE, nbase, k0, lane, 4 * HH), acc);
        if (t > 0) {
            #pragma unroll 4
            for (int k0 = 0; k0 < HH; k0 += 32)  // gates += h_{t-1} @ W_hh^T
                acc = wmma_bf16(loadA(hs, HH, (t - 1) * BB + mm * 16, k0, lane),
                                loadB(Whh, HH, nbase, k0, lane, 4 * HH), acc);
        }
        int half = lane >> 4, j = lane & 15;
        #pragma unroll
        for (int r = 0; r < 8; ++r)
            gl[g][mm * 16 + r + half * 8][j] = acc[r];
    }
    __syncthreads();

    for (int e = threadIdx.x; e < BB * 16; e += 256) {
        int m = e >> 4, j = e & 15;
        int n = n0 + j;
        float xi = gl[0][m][j] + bih[0 * HH + n] + bhh[0 * HH + n];
        float xf = gl[1][m][j] + bih[1 * HH + n] + bhh[1 * HH + n];
        float xg = gl[2][m][j] + bih[2 * HH + n] + bhh[2 * HH + n];
        float xo = gl[3][m][j] + bih[3 * HH + n] + bhh[3 * HH + n];
        float i_ = sigm(xi), f_ = sigm(xf), g_ = tanhf(xg), o_ = sigm(xo);
        float cp = (t == 0) ? 0.f : cbuf[m * HH + n];
        float cn = f_ * cp + i_ * g_;
        cbuf[m * HH + n] = cn;
        float h = o_ * tanhf(cn);
        hs[(size_t)(t * BB + m) * HH + n] = f2bf(h);
    }
}

// ---------------- output projection: out[b,t,v] = hs[t*64+b,:] . W_fc[v,:] + b_fc[v] ----------------
// Block = 8 waves; wave owns a 64x64 tile. grid.y = m-supertile (== t), grid.x = n-supertile of 512.
__global__ void fc_gemm(const unsigned short* __restrict__ hs, const unsigned short* __restrict__ Wfc,
                        const float* __restrict__ bfc, float* __restrict__ out) {
    int w = threadIdx.x >> 5, lane = threadIdx.x & 31;
    int mbase = blockIdx.y * 64;
    int nwave = blockIdx.x * 512 + w * 64;

    v8f acc[4][4];
    #pragma unroll
    for (int mm = 0; mm < 4; ++mm)
        #pragma unroll
        for (int nn = 0; nn < 4; ++nn) acc[mm][nn] = zero8();

    for (int k0 = 0; k0 < HH; k0 += 32) {
        v16bf a[4], b[4];
        #pragma unroll
        for (int mm = 0; mm < 4; ++mm) a[mm] = loadA(hs, HH, mbase + mm * 16, k0, lane);
        #pragma unroll
        for (int nn = 0; nn < 4; ++nn) b[nn] = loadB(Wfc, HH, nwave + nn * 16, k0, lane, VV);
        #pragma unroll
        for (int mm = 0; mm < 4; ++mm)
            #pragma unroll
            for (int nn = 0; nn < 4; ++nn)
                acc[mm][nn] = wmma_bf16(a[mm], b[nn], acc[mm][nn]);
    }

    int half = lane >> 4, nlo = lane & 15;
    #pragma unroll
    for (int nn = 0; nn < 4; ++nn) {
        int n = nwave + nn * 16 + nlo;
        if (n < VV) {
            float bias = bfc[n];
            #pragma unroll
            for (int mm = 0; mm < 4; ++mm)
                #pragma unroll
                for (int r = 0; r < 8; ++r) {
                    int row = mbase + mm * 16 + r + half * 8;   // row = t*64 + b
                    int t = row >> 6, b = row & 63;
                    out[(size_t)b * (TT * VV) + (size_t)t * VV + n] = acc[mm][nn][r] + bias;
                }
        }
    }
}

extern "C" void kernel_launch(void* const* d_in, const int* in_sizes, int n_in,
                              void* d_out, int out_size, void* d_ws, size_t ws_size,
                              hipStream_t stream) {
    const float* x       = (const float*)d_in[0];
    const int*   caption = (const int*)  d_in[1];
    const float* emb     = (const float*)d_in[2];
    const float* W_ih    = (const float*)d_in[3];
    const float* W_hh    = (const float*)d_in[4];
    const float* b_ih    = (const float*)d_in[5];
    const float* b_hh    = (const float*)d_in[6];
    const float* W_fc    = (const float*)d_in[7];
    const float* b_fc    = (const float*)d_in[8];
    float* out = (float*)d_out;

    char* ws = (char*)d_ws;
    size_t off = 0;
    unsigned short* xs   = (unsigned short*)(ws + off); off += (size_t)TT * BB * EE * 2;  // 3.28 MB
    unsigned short* hs   = (unsigned short*)(ws + off); off += (size_t)TT * BB * HH * 2;  // 3.28 MB
    float*          cbuf = (float*)         (ws + off); off += (size_t)BB * HH * 4;       // 128 KB
    unsigned short* Wihb = (unsigned short*)(ws + off); off += (size_t)4 * HH * EE * 2;   // 2 MB
    unsigned short* Whhb = (unsigned short*)(ws + off); off += (size_t)4 * HH * HH * 2;   // 2 MB
    unsigned short* Wfcb = (unsigned short*)(ws + off); off += (size_t)VV * HH * 2;       // 30.7 MB

    // Weight conversions f32 -> bf16
    {
        int n1 = 4 * HH * EE;
        cvt_k<<<(n1 + 255) / 256, 256, 0, stream>>>(W_ih, Wihb, n1);
        int n2 = 4 * HH * HH;
        cvt_k<<<(n2 + 255) / 256, 256, 0, stream>>>(W_hh, Whhb, n2);
        int n3 = VV * HH;
        cvt_k<<<(n3 + 255) / 256, 256, 0, stream>>>(W_fc, Wfcb, n3);
    }

    // Embedding gather + step-0 input
    {
        int n = TT * BB * EE;
        embed_k<<<(n + 255) / 256, 256, 0, stream>>>(x, caption, emb, xs);
    }

    // Sequential LSTM
    for (int t = 0; t < TT; ++t)
        lstm_step<<<HH / 16, 256, 0, stream>>>(xs, hs, cbuf, Wihb, Whhb, b_ih, b_hh, t);

    // Output projection + bias + [B,T,V] scatter
    {
        dim3 grid((VV + 511) / 512, (TT * BB) / 64);   // 59 x 50
        fc_gemm<<<grid, 256, 0, stream>>>(hs, Wfcb, b_fc, out);
    }
}
